// Convolution_56152402428536
// MI455X (gfx1250) — matrix operand order repaired and checked by reading, not compiled, not run on hardware
//
#include <hip/hip_runtime.h>
#include <hip/hip_bf16.h>
#include <math.h>

// ---------------------------------------------------------------------------
// e3nn-style scalar-irrep graph convolution for MI455X (gfx1250, wave32).
// Radial MLP (the 26-GFLOP hot spot) runs on v_wmma_f32_16x16x32_f16 and is
// fused with the per-edge message contraction + atomic scatter so the 819MB
// per-edge weight tensor never touches HBM.
// ---------------------------------------------------------------------------

typedef __attribute__((ext_vector_type(16))) _Float16 v16h;
typedef __attribute__((ext_vector_type(8)))  _Float16 v8h;
typedef __attribute__((ext_vector_type(8)))  float    v8f;
typedef __attribute__((ext_vector_type(4)))  float    v4f;

#define C_S        0.3826834323650898f   // sin(pi/8)
#define C_X        0.9238795325112867f   // cos(pi/8)
#define INV_SQRT8  0.35355339059327373f

static __device__ __forceinline__ unsigned pack2h(float lo, float hi) {
  _Float16 a = (_Float16)lo, b = (_Float16)hi;
  unsigned short ua, ub;
  __builtin_memcpy(&ua, &a, 2);
  __builtin_memcpy(&ub, &b, 2);
  return (unsigned)ua | ((unsigned)ub << 16);
}

// ---------------------------------------------------------------------------
// Pack fc_w1 [32x64] and fc_w2 [64x512] (f32, row-major) into the exact f16
// B-operand register image for V_WMMA_F32_16X16X32_F16 (wave32):
//   VGPR v, lanes 0-15 : K = 2v, 2v+1 (v<4) / 8+2v, 9+2v (v>=4), N = lane
//   VGPR v, lanes 16-31: K = 8+2v, 9+2v (v<4) / 16+2v, 17+2v (v>=4), N = lane-16
// w1p layout: [4 ntile][32 lane][8 dword]       (K=32, one WMMA step)
// w2p layout: [32 ntile][2 kstep][32 lane][8 dword]  (K=64 -> two steps)
// ---------------------------------------------------------------------------
__global__ __launch_bounds__(256) void pack_weights_kernel(
    const float* __restrict__ w1, const float* __restrict__ w2,
    unsigned* __restrict__ w1p, unsigned* __restrict__ w2p) {
  int id = blockIdx.x * 256 + threadIdx.x;
  if (id < 1024) {  // w1p
    int v = id & 7, lane = (id >> 3) & 31, t = id >> 8;
    int N  = t * 16 + (lane & 15);
    int k0 = 2 * v + 8 * (lane >> 4) + ((v >= 4) ? 8 : 0);
    w1p[id] = pack2h(w1[k0 * 64 + N], w1[(k0 + 1) * 64 + N]);
  }
  int id2 = id - 1024;
  if (id2 >= 0 && id2 < 16384) {  // w2p
    int v = id2 & 7, lane = (id2 >> 3) & 31, s = (id2 >> 8) & 1, t = id2 >> 9;
    int N  = t * 16 + (lane & 15);
    int k0 = 32 * s + 2 * v + 8 * (lane >> 4) + ((v >= 4) ? 8 : 0);
    w2p[id2] = pack2h(w2[k0 * 512 + N], w2[(k0 + 1) * 512 + N]);
  }
}

// ---------------------------------------------------------------------------
// Node prep: x[n,c,j] = sum_{i,a} node_input[n,c,i] * node_attr[n,a] * W_lin1[i,a,j]
// Stored as xg[n][j(8)][c(4)] so the edge kernel can stage rows with float4s.
// ---------------------------------------------------------------------------
__global__ __launch_bounds__(256) void node_prep_kernel(
    const float* __restrict__ node_input, const float* __restrict__ node_attr,
    const float* __restrict__ W_lin1, float* __restrict__ xg, int Nn) {
  __shared__ float wl[1024];  // W_lin1 [8][16][8]
  for (int i = threadIdx.x; i < 1024; i += 256) wl[i] = W_lin1[i];
  __syncthreads();
  int n = blockIdx.x * 256 + threadIdx.x;
  if (n >= Nn) return;
  float attr[16];
  #pragma unroll
  for (int a = 0; a < 16; ++a) attr[a] = node_attr[(size_t)n * 16 + a];
  float inp[32];
  #pragma unroll
  for (int q = 0; q < 32; ++q) inp[q] = node_input[(size_t)n * 32 + q];
  float wna[64];  // [i][j] = sum_a attr[a]*W[i][a][j]
  #pragma unroll
  for (int i = 0; i < 8; ++i)
    #pragma unroll
    for (int j = 0; j < 8; ++j) {
      float acc = 0.f;
      #pragma unroll
      for (int a = 0; a < 16; ++a) acc += attr[a] * wl[(i * 16 + a) * 8 + j];
      wna[i * 8 + j] = acc;
    }
  #pragma unroll
  for (int c = 0; c < 4; ++c)
    #pragma unroll
    for (int j = 0; j < 8; ++j) {
      float acc = 0.f;
      #pragma unroll
      for (int i = 0; i < 8; ++i) acc += inp[c * 8 + i] * wna[i * 8 + j];
      xg[(size_t)n * 32 + j * 4 + c] = acc;
    }
}

// ---------------------------------------------------------------------------
// Edge kernel: per wave, 16 edges.
//   h  = silu(feat @ W1 + b1)                 4x  v_wmma_f32_16x16x32_f16
//   ef = h @ W2 + b2   (consumed on the fly)  64x v_wmma_f32_16x16x32_f16
//   msg[c,d,o] = sum_i x[src][c,i]*ef[d,i,o]  VALU FMA vs LDS-staged x
//   agg[dst]  += msg                          global_atomic_add_f32
// ---------------------------------------------------------------------------
#define WAVES_PER_BLOCK 8

__global__ __launch_bounds__(256) void edge_kernel(
    const float* __restrict__ edge_features,
    const float* __restrict__ xg,
    const unsigned* __restrict__ w1p,
    const unsigned* __restrict__ w2p,
    const float* __restrict__ fc_b1,
    const float* __restrict__ fc_b2,
    const int* __restrict__ edge_src,
    const int* __restrict__ edge_dst,
    float* __restrict__ agg,
    int E_) {
  __shared__ _Float16 hlds[WAVES_PER_BLOCK][16][64];   // h, row-major f16
  __shared__ float    xlds[WAVES_PER_BLOCK][16][8][4]; // x[src] as [m][i][c]
  __shared__ float    b1s[64];
  __shared__ float    b2s[512];

  const int tid   = threadIdx.x;
  const int wv    = tid >> 5;
  const int l     = tid & 31;
  const int lo16  = l & 15;
  const int hi16  = l >> 4;        // which half-wave (A/B K-halves, C/D row-halves)
  const int mb    = hi16 * 8;      // base row (edge) this lane owns in C/D layout
  const int ihalf = (l >> 3) & 1;  // which i (DE) half within a 16-col tile
  const int oo    = l & 7;         // o (DO) index this lane owns

  if (tid < 64) b1s[tid] = fc_b1[tid];
  for (int i = tid; i < 512; i += 256) b2s[i] = fc_b2[i];

  const int e0 = (blockIdx.x * WAVES_PER_BLOCK + wv) * 16;

  // Stage x[src] rows for this wave's 16 edges (2 lanes per edge, float4s).
  {
    int m = l >> 1, p = l & 1;
    int ei = e0 + m; if (ei >= E_) ei = E_ - 1;
    int src = edge_src[ei];
    const float4* xr = (const float4*)(xg + (size_t)src * 32);
    float4* xd = (float4*)(&xlds[wv][m][0][0]);
    #pragma unroll
    for (int q = 0; q < 4; ++q) xd[p * 4 + q] = xr[p * 4 + q];
  }

  // Destination node ids for the 8 rows this lane owns.
  int dsts[8];
  #pragma unroll
  for (int r = 0; r < 8; ++r) {
    int ei = e0 + mb + r;
    dsts[r] = (ei < E_) ? edge_dst[ei] : -1;
  }

  __syncthreads();

  // ---- Layer 1: A operand straight from edge_features (native A layout):
  // lanes 0-15 hold K {0..7,16..23}, lanes 16-31 hold K {8..15,24..31}.
  v16h a1;
  {
    int row = e0 + lo16; if (row >= E_) row = E_ - 1;
    const float4* fp = (const float4*)(edge_features + (size_t)row * 32 + hi16 * 8);
    float4 f0 = fp[0], f1 = fp[1], f2 = fp[4], f3 = fp[5];
    a1[0]=(_Float16)f0.x;  a1[1]=(_Float16)f0.y;  a1[2]=(_Float16)f0.z;  a1[3]=(_Float16)f0.w;
    a1[4]=(_Float16)f1.x;  a1[5]=(_Float16)f1.y;  a1[6]=(_Float16)f1.z;  a1[7]=(_Float16)f1.w;
    a1[8]=(_Float16)f2.x;  a1[9]=(_Float16)f2.y;  a1[10]=(_Float16)f2.z; a1[11]=(_Float16)f2.w;
    a1[12]=(_Float16)f3.x; a1[13]=(_Float16)f3.y; a1[14]=(_Float16)f3.z; a1[15]=(_Float16)f3.w;
  }
  #pragma unroll
  for (int t = 0; t < 4; ++t) {
    v16h b = *(const v16h*)(w1p + (t * 32 + l) * 8);
    float bv = b1s[t * 16 + lo16];
    v8f cc = {bv, bv, bv, bv, bv, bv, bv, bv};           // bias via C operand
    v8f dd = __builtin_amdgcn_wmma_f32_16x16x32_f16(
        false, a1, false, b, (short)0, cc, false, false);
    #pragma unroll
    for (int r = 0; r < 8; ++r) {                        // SiLU, park in LDS
      float y = dd[r];
      // fast sigmoid: v_exp_f32 + v_rcp_f32 (avoid full IEEE divide chain)
      float hval = y * __builtin_amdgcn_rcpf(1.0f + __expf(-y));
      hlds[wv][mb + r][t * 16 + lo16] = (_Float16)hval;
    }
  }

  // Re-layout h as layer-2 A operands (K=64 -> two K=32 chunks) via LDS.
  v16h a2[2];
  #pragma unroll
  for (int s = 0; s < 2; ++s) {
    const v8h* hp = (const v8h*)(&hlds[wv][lo16][s * 32 + hi16 * 8]);
    v8h plo = hp[0];
    v8h phi = hp[2];  // +16 halves
    #pragma unroll
    for (int k = 0; k < 8; ++k) { a2[s][k] = plo[k]; a2[s][8 + k] = phi[k]; }
  }

  // ---- Layer 2 fused with message contraction + scatter.
  // 16-col tile t covers ef cols [16t,16t+16): d = t/4, i in {2(t%4), 2(t%4)+1},
  // o = 0..7; lane l holds (i = 2(t%4)+ihalf, o = oo) for rows mb..mb+7.
  #pragma unroll 1
  for (int d = 0; d < 8; ++d) {
    float macc[32];  // [c(4)][r(8)] partial messages for this (d, oo)
    #pragma unroll
    for (int q = 0; q < 32; ++q) macc[q] = 0.0f;
    #pragma unroll
    for (int tt = 0; tt < 4; ++tt) {
      int t = d * 4 + tt;
      float bv = b2s[t * 16 + lo16];
      v8f cc = {bv, bv, bv, bv, bv, bv, bv, bv};
      v16h b0 = *(const v16h*)(w2p + ((size_t)(t * 2 + 0) * 32 + l) * 8);
      v16h b1 = *(const v16h*)(w2p + ((size_t)(t * 2 + 1) * 32 + l) * 8);
      v8f ef = __builtin_amdgcn_wmma_f32_16x16x32_f16(
          false, a2[0], false, b0, (short)0, cc, false, false);
      ef = __builtin_amdgcn_wmma_f32_16x16x32_f16(
          false, a2[1], false, b1, (short)0, ef, false, false);
      int ii = 2 * tt + ihalf;
      #pragma unroll
      for (int r = 0; r < 8; ++r) {
        v4f xv = *(const v4f*)(&xlds[wv][mb + r][ii][0]);
        float e = ef[r];
        macc[0 * 8 + r] += xv[0] * e;
        macc[1 * 8 + r] += xv[1] * e;
        macc[2 * 8 + r] += xv[2] * e;
        macc[3 * 8 + r] += xv[3] * e;
      }
    }
    // Lane l^8 holds the other 4 of 8 i-terms for the same (edge, d, o).
    #pragma unroll
    for (int q = 0; q < 32; ++q) macc[q] += __shfl_xor(macc[q], 8, 32);
    // Split the scatter: ihalf==0 lanes write c=0,1; ihalf==1 lanes c=2,3.
    int c0 = ihalf * 2;
    #pragma unroll
    for (int r = 0; r < 8; ++r) {
      if (dsts[r] < 0) continue;
      size_t base = (size_t)dsts[r] * 256 + (size_t)d * 8 + oo; // [n][c][d][o]
      unsafeAtomicAdd(&agg[base + (size_t)c0 * 64],       macc[c0 * 8 + r]);
      unsafeAtomicAdd(&agg[base + (size_t)(c0 + 1) * 64], macc[(c0 + 1) * 8 + r]);
    }
  }
}

// ---------------------------------------------------------------------------
// Finalize (in-place on d_out, which holds agg):
//   out[n,c,d,p] = c_s * s[n,c,p] + (c_x/sqrt(8)) * sum_o agg[n,c,d,o]*W2na[o,p]
//   s[n,c,o]     = sum_{i,a} node_input[n,c,i] attr[n,a] W_sc[i,a,o]
// ---------------------------------------------------------------------------
__global__ __launch_bounds__(256) void finalize_kernel(
    const float* __restrict__ node_input, const float* __restrict__ node_attr,
    const float* __restrict__ W_sc, const float* __restrict__ W_lin2,
    float* __restrict__ out, int Nn) {
  __shared__ float wsc[1024], wl2[1024];
  for (int i = threadIdx.x; i < 1024; i += 256) { wsc[i] = W_sc[i]; wl2[i] = W_lin2[i]; }
  __syncthreads();
  int id = blockIdx.x * 256 + threadIdx.x;
  if (id >= Nn * 4) return;
  int n = id >> 2, c = id & 3;
  float attr[16];
  #pragma unroll
  for (int a = 0; a < 16; ++a) attr[a] = node_attr[(size_t)n * 16 + a];
  float inp[8];
  #pragma unroll
  for (int i = 0; i < 8; ++i) inp[i] = node_input[((size_t)n * 4 + c) * 8 + i];
  float s[8];
  #pragma unroll
  for (int o = 0; o < 8; ++o) s[o] = 0.f;
  #pragma unroll
  for (int i = 0; i < 8; ++i)
    #pragma unroll
    for (int o = 0; o < 8; ++o) {
      float acc = 0.f;
      #pragma unroll
      for (int a = 0; a < 16; ++a) acc += attr[a] * wsc[(i * 16 + a) * 8 + o];
      s[o] += inp[i] * acc;
    }
  float w2na[64];  // [o][p]
  #pragma unroll
  for (int o = 0; o < 8; ++o)
    #pragma unroll
    for (int p = 0; p < 8; ++p) {
      float acc = 0.f;
      #pragma unroll
      for (int a = 0; a < 16; ++a) acc += attr[a] * wl2[(o * 16 + a) * 8 + p];
      w2na[o * 8 + p] = acc;
    }
  const float scale = C_X * INV_SQRT8;
  float* base = out + (size_t)id * 64;
  #pragma unroll
  for (int d = 0; d < 8; ++d) {
    float a8[8];
    #pragma unroll
    for (int o = 0; o < 8; ++o) a8[o] = base[d * 8 + o];
    #pragma unroll
    for (int p = 0; p < 8; ++p) {
      float acc = 0.f;
      #pragma unroll
      for (int o = 0; o < 8; ++o) acc += a8[o] * w2na[o * 8 + p];
      base[d * 8 + p] = C_S * s[p] + scale * acc;
    }
  }
}

// ---------------------------------------------------------------------------
extern "C" void kernel_launch(void* const* d_in, const int* in_sizes, int n_in,
                              void* d_out, int out_size, void* d_ws, size_t ws_size,
                              hipStream_t stream) {
  const float* node_input    = (const float*)d_in[0];
  const float* node_attr     = (const float*)d_in[1];
  const float* edge_features = (const float*)d_in[2];
  const float* W_sc          = (const float*)d_in[3];
  const float* W_lin1        = (const float*)d_in[4];
  const float* W_lin2        = (const float*)d_in[5];
  const float* fc_w1         = (const float*)d_in[6];
  const float* fc_b1         = (const float*)d_in[7];
  const float* fc_w2         = (const float*)d_in[8];
  const float* fc_b2         = (const float*)d_in[9];
  const int*   edge_src      = (const int*)d_in[10];
  const int*   edge_dst      = (const int*)d_in[11];
  float*       out           = (float*)d_out;

  const int Nn = in_sizes[1] / 16;  // node_attr is [N,16]
  const int E_ = in_sizes[10];      // edge_src is [E]

  // Workspace: x [N][8][4] f32, then packed f16 weights (32B aligned).
  float*    xg  = (float*)d_ws;
  unsigned* w1p = (unsigned*)((char*)d_ws + (size_t)Nn * 32 * sizeof(float));
  unsigned* w2p = w1p + 1024;

  // d_out doubles as the scatter accumulator; zero it every call.
  hipMemsetAsync(d_out, 0, (size_t)out_size * sizeof(float), stream);

  pack_weights_kernel<<<(1024 + 16384 + 255) / 256, 256, 0, stream>>>(
      fc_w1, fc_w2, w1p, w2p);
  node_prep_kernel<<<(Nn + 255) / 256, 256, 0, stream>>>(
      node_input, node_attr, W_lin1, xg, Nn);
  edge_kernel<<<(E_ + 16 * WAVES_PER_BLOCK - 1) / (16 * WAVES_PER_BLOCK), 256, 0, stream>>>(
      edge_features, xg, w1p, w2p, fc_b1, fc_b2, edge_src, edge_dst, out, E_);
  finalize_kernel<<<(Nn * 4 + 255) / 256, 256, 0, stream>>>(
      node_input, node_attr, W_sc, W_lin2, out, Nn);
}
